// ConcatSquashLinearSA2_69020124446958
// MI455X (gfx1250) — matrix operand order repaired and checked by reading, not compiled, not run on hardware
//
#include <hip/hip_runtime.h>
#include <hip/hip_bf16.h>

// ---------------------------------------------------------------------------
// Problem: B=32, N=2048, DIM_IN=DIM_OUT=512, DIM_CTX=256
// out = ((x + x @ A_b) @ W_layer^T + b_layer) * gate_b + bias_b
// A_b = colnorm(softmax_rows(k_b outer k_b))
// Strategy: bf16x3 split GEMMs on v_wmma_f32_16x16x32_bf16 (memory-bound,
// ~fp32 accuracy), attention matrix materialized per batch as transposed
// hi/lo bf16 planes so WMMA B-fragments are contiguous loads.
// ---------------------------------------------------------------------------

typedef __bf16 v16bf __attribute__((ext_vector_type(16)));
typedef __bf16 bf16x8 __attribute__((ext_vector_type(8)));
typedef float  v8f   __attribute__((ext_vector_type(8)));

#define Bq 32
#define Nq 2048
#define D  512
#define Cq 256

// workspace byte offsets
#define K_OFF     (size_t)0
#define GATE_OFF  (size_t)(Bq*D*4)
#define BIAS_OFF  (size_t)(2*Bq*D*4)
#define MROW_OFF  (size_t)(3*Bq*D*4)
#define DROW_OFF  (size_t)(4*Bq*D*4)
#define CSUM_OFF  (size_t)(5*Bq*D*4)
#define WHI_OFF   (size_t)(6*Bq*D*4)
#define WLO_OFF   (WHI_OFF + (size_t)D*D*2)
#define AHI_OFF   (WLO_OFF + (size_t)D*D*2)
#define ALO_OFF   (AHI_OFF + (size_t)Bq*D*D*2)

// ---------------------------------------------------------------------------
// K1: gate / bias / k projections from ctx
// ---------------------------------------------------------------------------
__global__ __launch_bounds__(256)
void hyper_kernel(const float* __restrict__ ctx,
                  const float* __restrict__ Wg, const float* __restrict__ bg,
                  const float* __restrict__ Wb, const float* __restrict__ Wk,
                  float* __restrict__ k_ws, float* __restrict__ gate_ws,
                  float* __restrict__ bias_ws) {
  __shared__ float sc[Cq];
  const int b = blockIdx.x, t = threadIdx.x;
  sc[t] = ctx[b * Cq + t];
  __syncthreads();
  for (int o = t; o < D; o += 256) {
    const float* wg = Wg + o * Cq;
    const float* wb = Wb + o * Cq;
    const float* wk = Wk + o * Cq;
    float sg = 0.f, sb = 0.f, sk = 0.f;
    for (int c = 0; c < Cq; ++c) {
      float cv = sc[c];
      sg = fmaf(cv, wg[c], sg);
      sb = fmaf(cv, wb[c], sb);
      sk = fmaf(cv, wk[c], sk);
    }
    gate_ws[b * D + o] = 1.f / (1.f + __expf(-(sg + bg[o])));
    bias_ws[b * D + o] = sb;
    k_ws[b * D + o]    = sk;
  }
}

// ---------------------------------------------------------------------------
// K2a: per-batch softmax row stats (m_i, denom_i) + column sums
// energy[i][j] = k_i * k_j ; row max = k_i>=0 ? k_i*kmax : k_i*kmin
// ---------------------------------------------------------------------------
__global__ __launch_bounds__(256)
void attn_stats(const float* __restrict__ k_ws, float* __restrict__ mrow,
                float* __restrict__ drow, float* __restrict__ csum) {
  __shared__ float kv[D], mr[D], dr[D], red[256];
  const int b = blockIdx.x, t = threadIdx.x;
  kv[t]       = k_ws[b * D + t];
  kv[t + 256] = k_ws[b * D + t + 256];
  __syncthreads();
  red[t] = fmaxf(kv[t], kv[t + 256]);
  __syncthreads();
  for (int s = 128; s > 0; s >>= 1) { if (t < s) red[t] = fmaxf(red[t], red[t + s]); __syncthreads(); }
  const float kmax = red[0];
  __syncthreads();
  red[t] = fminf(kv[t], kv[t + 256]);
  __syncthreads();
  for (int s = 128; s > 0; s >>= 1) { if (t < s) red[t] = fminf(red[t], red[t + s]); __syncthreads(); }
  const float kmin = red[0];
  __syncthreads();
  for (int ii = 0; ii < 2; ++ii) {
    const int i = t + ii * 256;
    const float ki = kv[i];
    const float mi = (ki >= 0.f) ? ki * kmax : ki * kmin;
    float d = 0.f;
    for (int j = 0; j < D; ++j) d += __expf(ki * kv[j] - mi);
    mr[i] = mi; dr[i] = d;
    mrow[b * D + i] = mi; drow[b * D + i] = d;
  }
  __syncthreads();
  for (int jj = 0; jj < 2; ++jj) {
    const int j = t + jj * 256;
    const float kj = kv[j];
    float c = 0.f;
    for (int i = 0; i < D; ++i) c += __expf(kv[i] * kj - mr[i]) / dr[i];
    csum[b * D + j] = c;
  }
}

// ---------------------------------------------------------------------------
// K2b: materialize A^T[j][i] as hi/lo bf16 planes (K-contiguous for WMMA B)
// ---------------------------------------------------------------------------
__global__ __launch_bounds__(256)
void attn_planes(const float* __restrict__ k_ws, const float* __restrict__ mrow,
                 const float* __restrict__ drow, const float* __restrict__ csum,
                 __bf16* __restrict__ AhiT, __bf16* __restrict__ AloT) {
  __shared__ float kv[D], mr[D], dri[D], cinv[16];
  const int b = blockIdx.y, jb = blockIdx.x, t = threadIdx.x;
  for (int i = t; i < D; i += 256) {
    kv[i]  = k_ws[b * D + i];
    mr[i]  = mrow[b * D + i];
    dri[i] = 1.f / drow[b * D + i];
  }
  if (t < 16) cinv[t] = 1.f / (1e-9f + csum[b * D + jb * 16 + t]);
  __syncthreads();
  const size_t base = (size_t)b * D * D;
  for (int idx = t; idx < 16 * D; idx += 256) {
    const int jl = idx >> 9, i = idx & (D - 1);
    const int j = jb * 16 + jl;
    const float a = __expf(kv[i] * kv[j] - mr[i]) * dri[i] * cinv[jl];
    const __bf16 h = (__bf16)a;
    AhiT[base + (size_t)j * D + i] = h;
    AloT[base + (size_t)j * D + i] = (__bf16)(a - (float)h);
  }
}

// ---------------------------------------------------------------------------
// K3: W_layer[o][i] -> hi/lo bf16 planes (native layout is K-contiguous)
// ---------------------------------------------------------------------------
__global__ __launch_bounds__(256)
void conv_w(const float* __restrict__ W, __bf16* __restrict__ Whi,
            __bf16* __restrict__ Wlo) {
  const int idx = blockIdx.x * 256 + threadIdx.x;
  if (idx < D * D) {
    const float v = W[idx];
    const __bf16 h = (__bf16)v;
    Whi[idx] = h;
    Wlo[idx] = (__bf16)(v - (float)h);
  }
}

// ---------------------------------------------------------------------------
// WMMA fragment loaders (wave32; this code's contiguous k order, same for A and B, not the ISA table order)
// A 16x32: lane L (m=L%16): K = c0..c0+15 across VGPR0-7,
//          c0 = kb + (L<16 ? 0 : 16)
// B 32x16: lane L (n=L%16): K = k0..k0+15 across VGPR0-7,
//          k0 = kb + (L<16 ? 0 : 16)
// ---------------------------------------------------------------------------
__device__ __forceinline__ v16bf load_afrag(const __bf16* plane, int mrow_, int kb, int lane) {
  const int c0 = kb + ((lane & 16) ? 16 : 0);
  const __bf16* p = plane + mrow_ * D + c0;
  bf16x8 a0 = *(const bf16x8*)(p);
  bf16x8 a1 = *(const bf16x8*)(p + 8);
  return __builtin_shufflevector(a0, a1, 0, 1, 2, 3, 4, 5, 6, 7,
                                 8, 9, 10, 11, 12, 13, 14, 15);
}
__device__ __forceinline__ v16bf load_bfrag(const __bf16* plane, int nrow, int kb, int lane) {
  const int k0 = kb + ((lane & 16) ? 16 : 0);
  return *(const v16bf*)(plane + (size_t)nrow * D + k0);
}
__device__ __forceinline__ v8f wmma_bf16(v16bf a, v16bf b, v8f c) {
  return __builtin_amdgcn_wmma_f32_16x16x32_bf16(false, a, false, b,
                                                 (short)0, c, false, false);
}

// ---------------------------------------------------------------------------
// K4: fused  t = x + x@A  (stage 1, into LDS)  then
//            out = (t@W^T + b_layer)*gate + bias  (stage 2)
// 256 threads = 8 waves; 32 rows x 512 cols per block; bf16x3 split.
// ---------------------------------------------------------------------------
__global__ __launch_bounds__(256)
void gemm_fused(const float* __restrict__ x,
                const __bf16* __restrict__ AhiT, const __bf16* __restrict__ AloT,
                const __bf16* __restrict__ Whi,  const __bf16* __restrict__ Wlo,
                const float* __restrict__ b_layer,
                const float* __restrict__ gate_ws, const float* __restrict__ bias_ws,
                float* __restrict__ out) {
  extern __shared__ char smem[];
  __bf16* xhi = (__bf16*)smem;           // 32*512 bf16 = 32 KB
  __bf16* xlo = xhi + 32 * D;
  __bf16* thi = xlo + 32 * D;
  __bf16* tlo = thi + 32 * D;            // total 128 KB

  const int b    = blockIdx.y;
  const int r0   = blockIdx.x * 32;
  const int tid  = threadIdx.x;
  const int lane = tid & 31;
  const int wave = tid >> 5;

  // ---- load 32x512 x-tile, split into hi/lo bf16 ----
  const float4* xg = (const float4*)(x + ((size_t)b * Nq + r0) * D);
  for (int idx = tid; idx < 32 * (D / 4); idx += 256) {
    const float4 v = xg[idx];
    const int base = idx * 4;
    float f[4] = {v.x, v.y, v.z, v.w};
#pragma unroll
    for (int e = 0; e < 4; ++e) {
      const __bf16 h = (__bf16)f[e];
      xhi[base + e] = h;
      xlo[base + e] = (__bf16)(f[e] - (float)h);
    }
  }
  __syncthreads();

  const __bf16* Ahi_b = AhiT + (size_t)b * D * D;
  const __bf16* Alo_b = AloT + (size_t)b * D * D;
  const int mbase = (lane & 16) ? 8 : 0;

  // ---- stage 1: t = x + x @ A ----
  for (int s = 0; s < 8; ++s) {
    const int tile = wave + 8 * s;          // 64 tiles (2 x 32)
    const int nt = tile & 31, mt = tile >> 5;
    const int m_l = mt * 16 + (lane & 15);  // A-fragment row (local)
    const int n_g = nt * 16 + (lane & 15);  // B-fragment column j
    v8f acc = {};
    for (int kb = 0; kb < D; kb += 32) {
      v16bf ah = load_afrag(xhi, m_l, kb, lane);
      v16bf al = load_afrag(xlo, m_l, kb, lane);
      v16bf bh = load_bfrag(Ahi_b, n_g, kb, lane);
      v16bf bl = load_bfrag(Alo_b, n_g, kb, lane);
      acc = wmma_bf16(ah, bh, acc);
      acc = wmma_bf16(ah, bl, acc);
      acc = wmma_bf16(al, bh, acc);
    }
    const int cl = nt * 16 + (lane & 15);   // C/D: N = lane%16
#pragma unroll
    for (int r = 0; r < 8; ++r) {           // C/D: M = r + (lane>=16 ? 8:0)
      const int ml = mt * 16 + mbase + r;
      const float xv = (float)xhi[ml * D + cl] + (float)xlo[ml * D + cl];
      const float t = acc[r] + xv;
      const __bf16 h = (__bf16)t;
      thi[ml * D + cl] = h;
      tlo[ml * D + cl] = (__bf16)(t - (float)h);
    }
  }
  __syncthreads();

  // ---- stage 2: out = (t @ W^T + b_layer) * gate + bias ----
  for (int s = 0; s < 8; ++s) {
    const int tile = wave + 8 * s;
    const int nt = tile & 31, mt = tile >> 5;
    const int m_l = mt * 16 + (lane & 15);
    const int o_g = nt * 16 + (lane & 15);  // output channel o
    v8f acc = {};
    for (int kb = 0; kb < D; kb += 32) {
      v16bf ah = load_afrag(thi, m_l, kb, lane);
      v16bf al = load_afrag(tlo, m_l, kb, lane);
      v16bf bh = load_bfrag(Whi, o_g, kb, lane);
      v16bf bl = load_bfrag(Wlo, o_g, kb, lane);
      acc = wmma_bf16(ah, bh, acc);
      acc = wmma_bf16(ah, bl, acc);
      acc = wmma_bf16(al, bh, acc);
    }
    const float blv = b_layer[o_g];
    const float g   = gate_ws[b * D + o_g];
    const float bi  = bias_ws[b * D + o_g];
#pragma unroll
    for (int r = 0; r < 8; ++r) {
      const int row = r0 + mt * 16 + mbase + r;
      out[((size_t)b * Nq + row) * D + o_g] = (acc[r] + blv) * g + bi;
    }
  }
}

// ---------------------------------------------------------------------------
extern "C" void kernel_launch(void* const* d_in, const int* in_sizes, int n_in,
                              void* d_out, int out_size, void* d_ws, size_t ws_size,
                              hipStream_t stream) {
  const float* ctx     = (const float*)d_in[0];
  const float* x       = (const float*)d_in[1];
  const float* W_layer = (const float*)d_in[2];
  const float* b_layer = (const float*)d_in[3];
  const float* W_bias  = (const float*)d_in[4];
  const float* W_gate  = (const float*)d_in[5];
  const float* b_gate  = (const float*)d_in[6];
  const float* W_k     = (const float*)d_in[7];
  float* out = (float*)d_out;
  char*  ws  = (char*)d_ws;

  float* k_ws    = (float*)(ws + K_OFF);
  float* gate_ws = (float*)(ws + GATE_OFF);
  float* bias_ws = (float*)(ws + BIAS_OFF);
  float* mrow    = (float*)(ws + MROW_OFF);
  float* drow    = (float*)(ws + DROW_OFF);
  float* csum    = (float*)(ws + CSUM_OFF);
  __bf16* Whi    = (__bf16*)(ws + WHI_OFF);
  __bf16* Wlo    = (__bf16*)(ws + WLO_OFF);
  __bf16* AhiT   = (__bf16*)(ws + AHI_OFF);
  __bf16* AloT   = (__bf16*)(ws + ALO_OFF);

  hyper_kernel<<<Bq, 256, 0, stream>>>(ctx, W_gate, b_gate, W_bias, W_k,
                                       k_ws, gate_ws, bias_ws);
  attn_stats<<<Bq, 256, 0, stream>>>(k_ws, mrow, drow, csum);
  attn_planes<<<dim3(32, Bq), 256, 0, stream>>>(k_ws, mrow, drow, csum, AhiT, AloT);
  conv_w<<<(D * D + 255) / 256, 256, 0, stream>>>(W_layer, Whi, Wlo);
  gemm_fused<<<dim3(Nq / 32, Bq), 256, 128 * 1024, stream>>>(
      x, AhiT, AloT, Whi, Wlo, b_layer, gate_ws, bias_ws, out);
}
